// SCANModel_40699110097542
// MI455X (gfx1250) — matrix operand order repaired and checked by reading, not compiled, not run on hardware
//
#include <hip/hip_runtime.h>

// ---------------------------------------------------------------------------
// SCAN GRU decoder for MI455X (gfx1250): bf16 WMMA (v_wmma_f32_16x16x32_bf16)
// GEMMs with f32 state, stream-ordered 32-step autoregressive loop.
// Round 2: packed [6H,H] gate weights -> single base pointer + full K unroll
// so all B-frag addresses fold into 24-bit instruction immediates (no u64
// adds, lower VGPR pressure, loads overlap WMMAs instead of serializing).
// ---------------------------------------------------------------------------

#define DEV static __device__ __forceinline__

constexpr int B      = 256;
constexpr int H      = 512;
constexpr int V      = 16000;
constexpr int MSTEP  = 32;
constexpr int SOS_TK = 1;
constexpr int EOS_TK = 2;
constexpr int NT_V   = V / 16;   // 1000 fc column tiles
constexpr int LDT    = 520;      // padded LDS row stride in u16 (16B-aligned rows)
constexpr int GSTRIDE = H * H;   // elements between gate blocks in packed W

typedef __attribute__((ext_vector_type(16))) __bf16 v16bf;
typedef __attribute__((ext_vector_type(8)))  float  v8f;

union Frag {
  v16bf v;
  uint4 q[2];
};

DEV unsigned short f2bf(float f) {          // round-to-nearest-even f32 -> bf16
  unsigned int u = __float_as_uint(f);
  u += 0x7fffu + ((u >> 16) & 1u);
  return (unsigned short)(u >> 16);
}

DEV float sigmoidf(float x) { return 1.0f / (1.0f + __expf(-x)); }

DEV v8f wmma_bf16(const Frag& a, const Frag& b, v8f c) {
  return __builtin_amdgcn_wmma_f32_16x16x32_bf16(
      /*neg_a=*/false, a.v, /*neg_b=*/false, b.v,
      /*c_mod=*/(short)0, c, /*reuse_a=*/false, /*reuse_b=*/false);
}

DEV Frag ld_frag_g(const unsigned short* p) {   // two 8-elem K chunks from global
  Frag f;
  f.q[0] = *(const uint4*)p;
  f.q[1] = *(const uint4*)(p + 16);
  return f;
}

DEV Frag ld_frag_lds(const unsigned short* p) { // two 8-elem K chunks from LDS
  Frag f;
  f.q[0] = *(const uint4*)p;
  f.q[1] = *(const uint4*)(p + 16);
  return f;
}

// ---------------------------------------------------------------------------
// Weight conversion f32 -> bf16 (once per launch; weights then live in L2)
// ---------------------------------------------------------------------------
__global__ void cvt_bf16_kernel(const float* __restrict__ src,
                                unsigned short* __restrict__ dst, int n) {
  int i = blockIdx.x * blockDim.x + threadIdx.x;
  if (i < n) dst[i] = f2bf(src[i]);
}

// ---------------------------------------------------------------------------
// State init: h0 = h1 = sem_f (f32 + bf16 copies), inputs = SOS, finished = 0
// ---------------------------------------------------------------------------
__global__ void init_kernel(const float* __restrict__ sem,
                            unsigned short* __restrict__ h0b,
                            unsigned short* __restrict__ h1b,
                            float* __restrict__ h0f, float* __restrict__ h1f,
                            int* __restrict__ inputs, int* __restrict__ finished) {
  int gid = blockIdx.x * blockDim.x + threadIdx.x;
  if (gid < B * H) {
    float x = sem[gid];
    h0f[gid] = x; h1f[gid] = x;
    unsigned short bf = f2bf(x);
    h0b[gid] = bf; h1b[gid] = bf;
  }
  if (gid < B) { inputs[gid] = SOS_TK; finished[gid] = 0; }
}

// ---------------------------------------------------------------------------
// x = bf16(emb[inputs])  (one block per batch row)
// ---------------------------------------------------------------------------
__global__ __launch_bounds__(128) void gather_kernel(
    const float* __restrict__ emb, const int* __restrict__ inputs,
    unsigned short* __restrict__ x0) {
  int b = blockIdx.x;
  const float* src = emb + (size_t)inputs[b] * H;
  unsigned short* dst = x0 + b * H;
  for (int c = threadIdx.x; c < H; c += 128) dst[c] = f2bf(src[c]);
}

// ---------------------------------------------------------------------------
// Fused GRU cell. W packed [6H, H] bf16: rows [0,3H) = W_ih, [3H,6H) = W_hh.
// All six gate B-fragments addressed at constant offsets g*GSTRIDE from one
// per-lane base; K loop fully unrolled -> immediate-offset global loads.
// grid = (B/16, H/64), block = 128 (4 waves; wave -> one 16-col tile)
// ---------------------------------------------------------------------------
__global__ __launch_bounds__(128) void gru_kernel(
    const unsigned short* __restrict__ Ax,    // [B,H] bf16 input x
    const unsigned short* __restrict__ Ah,    // [B,H] bf16 h_prev
    const unsigned short* __restrict__ W,     // [6H,H] bf16 packed ih|hh
    const float* __restrict__ bih, const float* __restrict__ bhh,
    const float* __restrict__ holdf,          // [B,H] f32 h_prev
    const int* __restrict__ finished,
    float* __restrict__ hnewf,                // [B,H] f32 frozen h_new
    unsigned short* __restrict__ hnewbf,      // [B,H] bf16 frozen h_new
    unsigned short* __restrict__ xout)        // [B,H] bf16 unfrozen h_n (or null)
{
  __shared__ unsigned short xs[16 * LDT];
  __shared__ unsigned short hs[16 * LDT];
  const int tid   = threadIdx.x;
  const int bbase = blockIdx.x * 16;

  // Stage 16x512 bf16 A-tiles (x and h) into padded LDS, b128 per thread.
  for (int i = tid; i < 1024; i += 128) {       // 64 uint4 per row
    int row = i >> 6, c = i & 63;
    *((uint4*)(xs + row * LDT) + c) = *((const uint4*)(Ax + (bbase + row) * H) + c);
    *((uint4*)(hs + row * LDT) + c) = *((const uint4*)(Ah + (bbase + row) * H) + c);
  }
  __syncthreads();

  const int wave = tid >> 5;
  const int lane = tid & 31;
  const int n    = lane & 15;
  const int half = lane >> 4;
  const int colbase = (blockIdx.y * 4 + wave) * 16;
  const int col  = colbase + n;

  // Single per-lane bases; every other address is base + compile-time const.
  const unsigned short* wb = W + (size_t)col * H + 8 * half;   // gi_r row
  const unsigned short* xb = xs + n * LDT + 8 * half;
  const unsigned short* hb = hs + n * LDT + 8 * half;

  v8f zero = {0.f, 0.f, 0.f, 0.f, 0.f, 0.f, 0.f, 0.f};
  v8f a0 = zero, a1 = zero, a2 = zero, a3 = zero, a4 = zero, a5 = zero;

#pragma unroll
  for (int k0 = 0; k0 < H; k0 += 32) {
    Frag ax = ld_frag_lds(xb + k0);
    Frag ah = ld_frag_lds(hb + k0);
    Frag b0 = ld_frag_g(wb + k0 + 0 * GSTRIDE);   // gi_r
    Frag b1 = ld_frag_g(wb + k0 + 1 * GSTRIDE);   // gi_z
    Frag b2 = ld_frag_g(wb + k0 + 2 * GSTRIDE);   // gi_n
    Frag b3 = ld_frag_g(wb + k0 + 3 * GSTRIDE);   // gh_r
    Frag b4 = ld_frag_g(wb + k0 + 4 * GSTRIDE);   // gh_z
    Frag b5 = ld_frag_g(wb + k0 + 5 * GSTRIDE);   // gh_n
    a0 = wmma_bf16(ax, b0, a0);
    a1 = wmma_bf16(ax, b1, a1);
    a2 = wmma_bf16(ax, b2, a2);
    a3 = wmma_bf16(ah, b3, a3);
    a4 = wmma_bf16(ah, b4, a4);
    a5 = wmma_bf16(ah, b5, a5);
  }

  const float v_bir = bih[col], v_biz = bih[H + col], v_bin = bih[2 * H + col];
  const float v_bhr = bhh[col], v_bhz = bhh[H + col], v_bhn = bhh[2 * H + col];

#pragma unroll
  for (int v = 0; v < 8; ++v) {
    int row   = bbase + v + 8 * half;       // C/D layout: M = vgpr + 8*(lane/16)
    float hold = holdf[row * H + col];
    int   fin  = finished[row];
    float r  = sigmoidf(a0[v] + v_bir + a3[v] + v_bhr);
    float z  = sigmoidf(a1[v] + v_biz + a4[v] + v_bhz);
    float nn = tanhf(a2[v] + v_bin + r * (a5[v] + v_bhn));
    float hn = (1.0f - z) * nn + z * hold;
    float hw = fin ? hold : hn;             // freeze finished rows
    hnewf[row * H + col]  = hw;
    hnewbf[row * H + col] = f2bf(hw);
    if (xout) xout[row * H + col] = f2bf(hn);   // unfrozen h0n feeds layer 1
  }
}

// ---------------------------------------------------------------------------
// fc: out = h1 @ fc_W^T + fc_b ; write masked logits ; per-tile argmax partial
// grid = (B/16, NT_V/8), block = 256 (8 waves; wave -> one 16-col V tile)
// ---------------------------------------------------------------------------
__global__ __launch_bounds__(256) void fc_kernel(
    const unsigned short* __restrict__ A,     // [B,H] bf16 h1 (frozen)
    const unsigned short* __restrict__ Wfc,   // [V,H] bf16
    const float* __restrict__ bfc,
    const int* __restrict__ finished,
    float* __restrict__ logits,               // base of [B,M,V] region in d_out
    float* __restrict__ pval, int* __restrict__ pidx,
    int t)
{
  __shared__ unsigned short as[16 * LDT];
  const int tid   = threadIdx.x;
  const int bbase = blockIdx.x * 16;

  for (int i = tid; i < 1024; i += 256) {
    int row = i >> 6, c = i & 63;
    *((uint4*)(as + row * LDT) + c) = *((const uint4*)(A + (bbase + row) * H) + c);
  }
  __syncthreads();

  const int wave = tid >> 5;
  const int lane = tid & 31;
  const int n    = lane & 15;
  const int half = lane >> 4;
  const int tile = blockIdx.y * 8 + wave;     // 0..999
  const int col  = tile * 16 + n;

  const unsigned short* wr = Wfc + (size_t)col * H + 8 * half;
  const unsigned short* ab = as + n * LDT + 8 * half;

  v8f acc = {0.f, 0.f, 0.f, 0.f, 0.f, 0.f, 0.f, 0.f};
#pragma unroll
  for (int k0 = 0; k0 < H; k0 += 32) {
    Frag a = ld_frag_lds(ab + k0);
    Frag b = ld_frag_g(wr + k0);
    acc = wmma_bf16(a, b, acc);
  }

  const float bias = bfc[col];
#pragma unroll
  for (int v = 0; v < 8; ++v) {
    int   row  = bbase + v + 8 * half;
    float outv = acc[v] + bias;
    int   un   = !finished[row];
    logits[(size_t)(row * MSTEP + t) * V + col] = un ? outv : 0.0f;

    // first-max argmax across the 16 lanes (columns) holding this row
    float bv = outv; int bi = col;
#pragma unroll
    for (int m = 1; m < 16; m <<= 1) {
      float ov = __shfl_xor(bv, m, 32);
      int   oi = __shfl_xor(bi, m, 32);
      if (ov > bv || (ov == bv && oi < bi)) { bv = ov; bi = oi; }
    }
    if (n == 0) {
      pval[row * NT_V + tile] = bv;
      pidx[row * NT_V + tile] = bi;
    }
  }
}

// ---------------------------------------------------------------------------
// Final argmax over 1000 tiles per row + decode/EOS state update
// ---------------------------------------------------------------------------
__global__ __launch_bounds__(128) void argmax_update_kernel(
    const float* __restrict__ pval, const int* __restrict__ pidx,
    int* __restrict__ inputs, int* __restrict__ finished,
    float* __restrict__ dec_out, float* __restrict__ mask_out, int t)
{
  __shared__ float sv[128];
  __shared__ int   si[128];
  const int b   = blockIdx.x;
  const int tid = threadIdx.x;

  float bv = -3.402823466e38f;
  int   bi = 0x7fffffff;
  for (int i = tid; i < NT_V; i += 128) {
    float v = pval[b * NT_V + i];
    int   x = pidx[b * NT_V + i];
    if (v > bv || (v == bv && x < bi)) { bv = v; bi = x; }
  }
  sv[tid] = bv; si[tid] = bi;
  __syncthreads();
  for (int s = 64; s > 0; s >>= 1) {
    if (tid < s) {
      float ov = sv[tid + s]; int oi = si[tid + s];
      if (ov > sv[tid] || (ov == sv[tid] && oi < si[tid])) { sv[tid] = ov; si[tid] = oi; }
    }
    __syncthreads();
  }
  if (tid == 0) {
    int dec = si[0];
    int un  = !finished[b];
    dec_out[b * MSTEP + t]  = un ? (float)dec : -1.0f;
    mask_out[b * MSTEP + t] = un ? 1.0f : 0.0f;
    if (un) {
      inputs[b] = dec;
      if (dec == EOS_TK) finished[b] = 1;
    }
  }
}

// ---------------------------------------------------------------------------
extern "C" void kernel_launch(void* const* d_in, const int* in_sizes, int n_in,
                              void* d_out, int out_size, void* d_ws, size_t ws_size,
                              hipStream_t stream) {
  (void)in_sizes; (void)n_in; (void)out_size; (void)ws_size;

  const float* sem  = (const float*)d_in[0];
  const float* emb  = (const float*)d_in[1];
  const float* Wih0 = (const float*)d_in[2];
  const float* Whh0 = (const float*)d_in[3];
  const float* bih0 = (const float*)d_in[4];
  const float* bhh0 = (const float*)d_in[5];
  const float* Wih1 = (const float*)d_in[6];
  const float* Whh1 = (const float*)d_in[7];
  const float* bih1 = (const float*)d_in[8];
  const float* bhh1 = (const float*)d_in[9];
  const float* fcW  = (const float*)d_in[10];
  const float* fcb  = (const float*)d_in[11];

  char* ws = (char*)d_ws;
  size_t off = 0;
  auto take = [&](size_t bytes) -> char* {
    char* p = ws + off;
    off = (off + bytes + 255) & ~(size_t)255;
    return p;
  };

  // Packed [6H,H] per-layer gate weights: rows [0,3H)=W_ih, [3H,6H)=W_hh.
  unsigned short* w0b = (unsigned short*)take((size_t)6 * H * H * 2);
  unsigned short* w1b = (unsigned short*)take((size_t)6 * H * H * 2);
  unsigned short* fcwb  = (unsigned short*)take((size_t)V * H * 2);
  unsigned short* x0    = (unsigned short*)take((size_t)B * H * 2);
  unsigned short* x1    = (unsigned short*)take((size_t)B * H * 2);
  unsigned short* h0bf[2] = {(unsigned short*)take((size_t)B * H * 2),
                             (unsigned short*)take((size_t)B * H * 2)};
  unsigned short* h1bf[2] = {(unsigned short*)take((size_t)B * H * 2),
                             (unsigned short*)take((size_t)B * H * 2)};
  float* h0f[2] = {(float*)take((size_t)B * H * 4), (float*)take((size_t)B * H * 4)};
  float* h1f[2] = {(float*)take((size_t)B * H * 4), (float*)take((size_t)B * H * 4)};
  int*   inputs   = (int*)take((size_t)B * 4);
  int*   finished = (int*)take((size_t)B * 4);
  float* pval     = (float*)take((size_t)B * NT_V * 4);
  int*   pidx     = (int*)take((size_t)B * NT_V * 4);

  float* dec_out  = (float*)d_out;                       // [B,M] decoded
  float* logits   = dec_out + (size_t)B * MSTEP;         // [B,M,V]
  float* mask_out = logits + (size_t)B * MSTEP * V;      // [B,M] mask

  // Convert all weights to bf16 (L2-resident for the whole decode).
  const int NG = 3 * H * H;
  cvt_bf16_kernel<<<(NG + 255) / 256, 256, 0, stream>>>(Wih0, w0b, NG);
  cvt_bf16_kernel<<<(NG + 255) / 256, 256, 0, stream>>>(Whh0, w0b + NG, NG);
  cvt_bf16_kernel<<<(NG + 255) / 256, 256, 0, stream>>>(Wih1, w1b, NG);
  cvt_bf16_kernel<<<(NG + 255) / 256, 256, 0, stream>>>(Whh1, w1b + NG, NG);
  const int NF = V * H;
  cvt_bf16_kernel<<<(NF + 255) / 256, 256, 0, stream>>>(fcW, fcwb, NF);

  init_kernel<<<(B * H + 255) / 256, 256, 0, stream>>>(
      sem, h0bf[0], h1bf[0], h0f[0], h1f[0], inputs, finished);

  for (int t = 0; t < MSTEP; ++t) {
    const int cur = t & 1, nxt = cur ^ 1;
    gather_kernel<<<B, 128, 0, stream>>>(emb, inputs, x0);
    gru_kernel<<<dim3(B / 16, H / 64), 128, 0, stream>>>(
        x0, h0bf[cur], w0b, bih0, bhh0, h0f[cur], finished,
        h0f[nxt], h0bf[nxt], x1);
    gru_kernel<<<dim3(B / 16, H / 64), 128, 0, stream>>>(
        x1, h1bf[cur], w1b, bih1, bhh1, h1f[cur], finished,
        h1f[nxt], h1bf[nxt], nullptr);
    fc_kernel<<<dim3(B / 16, NT_V / 8), 256, 0, stream>>>(
        h1bf[nxt], fcwb, fcb, finished, logits, pval, pidx, t);
    argmax_update_kernel<<<B, 128, 0, stream>>>(
        pval, pidx, inputs, finished, dec_out, mask_out, t);
  }
}
